// CustomTransformerDecoderLayer_3753801416834
// MI455X (gfx1250) — matrix operand order repaired and checked by reading, not compiled, not run on hardware
//
#include <hip/hip_runtime.h>
#include <hip/hip_bf16.h>
#include <math.h>

typedef __attribute__((ext_vector_type(16))) __bf16    v16bf;
typedef __attribute__((ext_vector_type(8)))  float     v8f;
typedef __attribute__((ext_vector_type(8)))  unsigned  v8u;

// ---------------------------------------------------------------------------
// bf16 conversion helpers. Round-half-up (+0x8000, truncate): differs from RNE
// only on exact 16-bit ties; 2 v_add + 1 perm-style pack per float pair.
// ---------------------------------------------------------------------------
__device__ __forceinline__ unsigned pk2bf(float lo, float hi) {
  const unsigned a = __builtin_bit_cast(unsigned, lo) + 0x8000u;
  const unsigned b = __builtin_bit_cast(unsigned, hi) + 0x8000u;
  return (a >> 16) | (b & 0xffff0000u);   // {hi.bf16, lo.bf16} packed -> v_perm_b32
}
__device__ __forceinline__ __bf16 f2bf1(float f) {
  const unsigned a = __builtin_bit_cast(unsigned, f) + 0x8000u;
  const unsigned short h = (unsigned short)(a >> 16);
  return __builtin_bit_cast(__bf16, h);
}

// A fragment (16-bit A layout, ISA 7.12.2): lane holds row (lane&15),
// K = {k0+h*8..+7} U {k0+16+h*8..+7}; pa points at Arow + k0 + h*8.
__device__ __forceinline__ v16bf ldfragA(const float* __restrict__ pa) {
  const float4 a0 = *(const float4*)(pa);
  const float4 a1 = *(const float4*)(pa + 4);
  const float4 a2 = *(const float4*)(pa + 16);
  const float4 a3 = *(const float4*)(pa + 20);
  v8u u;
  u[0] = pk2bf(a0.x, a0.y); u[1] = pk2bf(a0.z, a0.w);
  u[2] = pk2bf(a1.x, a1.y); u[3] = pk2bf(a1.z, a1.w);
  u[4] = pk2bf(a2.x, a2.y); u[5] = pk2bf(a2.z, a2.w);
  u[6] = pk2bf(a3.x, a3.y); u[7] = pk2bf(a3.z, a3.w);
  return __builtin_bit_cast(v16bf, u);
}
// B fragment: lane holds column n0+(lane&15), K = {k0+h*8..+7} U {k0+16+h*8..+7} (stride N).
__device__ __forceinline__ v16bf ldfragB(const float* __restrict__ pb, int N) {
  v8u u;
  #pragma unroll
  for (int i = 0; i < 8; ++i) {
    const int k = (i < 4) ? 2 * i : 2 * i + 8;
    u[i] = pk2bf(pb[(size_t)k * N], pb[(size_t)(k + 1) * N]);
  }
  return __builtin_bit_cast(v16bf, u);
}

// ---------------------------------------------------------------------------
// Generic GEMM: C[M,N] = act(A[M,K] @ W[K,N] + bias), fp32 in/out, bf16 WMMA.
// One wave owns one 16x16 output tile; K stepped 32 per v_wmma_f32_16x16x32_bf16.
// M multiple of 16, K multiple of 128 (true for all call sites). N may be ragged.
// ---------------------------------------------------------------------------
__global__ void gemm_bf16_wmma(const float* __restrict__ A,
                               const float* __restrict__ W,
                               const float* __restrict__ bias,
                               float* __restrict__ C,
                               int M, int N, int K, int relu) {
  const int lane = threadIdx.x & 31;
  const int wid  = threadIdx.x >> 5;
  const int ntN  = (N + 15) >> 4;
  const int ntiles = (M >> 4) * ntN;
  const int tile = blockIdx.x * 8 + wid;
  if (tile >= ntiles) return;                 // wave-uniform: EXEC stays all-ones
  const int tM = tile / ntN;
  const int n0 = (tile % ntN) << 4;
  const int half = lane >> 4;
  const int rowA = (tM << 4) + (lane & 15);
  const float* Arow = A + (size_t)rowA * K + half * 8;
  const float* Bcol = W + (size_t)(half * 8) * N + n0 + (lane & 15);

  v8f acc = {0.f, 0.f, 0.f, 0.f, 0.f, 0.f, 0.f, 0.f};

  if (n0 + 16 <= N) {                         // full-width tile (uniform branch)
    #pragma unroll 4
    for (int k0 = 0; k0 < K; k0 += 32) {
      const v16bf a = ldfragA(Arow + k0);
      const v16bf b = ldfragB(Bcol + (size_t)k0 * N, N);
      acc = __builtin_amdgcn_wmma_f32_16x16x32_bf16(false, a, false, b,
                                                    (short)0, acc, false, false);
    }
  } else {                                    // ragged N tail (reg head N=24)
    for (int k0 = 0; k0 < K; k0 += 32) {
      const v16bf a = ldfragA(Arow + k0);
      const float* pb = Bcol + (size_t)k0 * N;
      v16bf b;
      #pragma unroll
      for (int j = 0; j < 16; ++j) b[j] = (n0 + (lane & 15) < N) ? f2bf1(pb[(size_t)(j < 8 ? j : j + 8) * N]) : f2bf1(0.0f);
      acc = __builtin_amdgcn_wmma_f32_16x16x32_bf16(false, a, false, b,
                                                    (short)0, acc, false, false);
    }
  }

  const int col = n0 + (lane & 15);
  if (col < N) {
    const float bv = bias ? bias[col] : 0.0f;
    #pragma unroll
    for (int r = 0; r < 8; ++r) {
      const int orow = (tM << 4) + r + half * 8;
      float v = acc[r] + bv;
      if (relu) v = fmaxf(v, 0.0f);
      C[(size_t)orow * N + col] = v;
    }
  }
}

// ---------------------------------------------------------------------------
// Fused: pt_w softmax + bilinear corner weights + weighted channel gather.
// One block per (b,m) token; thread = channel. Emits h[row,256] and s[row].
// ---------------------------------------------------------------------------
__global__ void bev_gather_kernel(const float* __restrict__ traj,
                                  const float* __restrict__ noisy,
                                  const float* __restrict__ bev,
                                  const float* __restrict__ attn_w,
                                  const float* __restrict__ attn_b,
                                  float* __restrict__ h, float* __restrict__ s) {
  const int row = blockIdx.x;           // b*20 + m
  const int b   = row / 20;
  const int tid = threadIdx.x;          // channel 0..255
  const int lane = tid & 31, wid = tid >> 5;
  __shared__ float slog[8];
  __shared__ float sptw[8];

  const float* trow = traj + (size_t)row * 256;
  float part = 0.f;
  for (int c = lane; c < 256; c += 32) part += trow[c] * attn_w[c * 8 + wid];
  for (int off = 16; off; off >>= 1) part += __shfl_xor(part, off, 32);
  if (lane == 0) slog[wid] = part + attn_b[wid];
  __syncthreads();
  if (tid == 0) {
    float mx = slog[0];
    for (int p = 1; p < 8; ++p) mx = fmaxf(mx, slog[p]);
    float e[8], se = 0.f;
    for (int p = 0; p < 8; ++p) { e[p] = __expf(slog[p] - mx); se += e[p]; }
    for (int p = 0; p < 8; ++p) sptw[p] = e[p] / se;
  }
  __syncthreads();

  const float* np = noisy + (size_t)row * 16;                 // 8 points * (x,y)
  const float* fb = bev + ((size_t)b * 256 + tid) * 16384;    // channel plane 128x128
  float acc = 0.f, stot = 0.f;
  for (int p = 0; p < 8; ++p) {
    const float ptw = sptw[p];
    const float gx = np[p * 2 + 0] * (1.f / 32.f);
    const float gy = np[p * 2 + 1] * (1.f / 32.f);
    const float x = (gx + 1.f) * 64.f - 0.5f;
    const float y = (gy + 1.f) * 64.f - 0.5f;
    const float x0f = floorf(x), y0f = floorf(y);
    const int x0 = (int)x0f, y0 = (int)y0f;
    const float fx = x - x0f, fy = y - y0f;
    const float w00 = (1.f - fx) * (1.f - fy), w10 = fx * (1.f - fy);
    const float w01 = (1.f - fx) * fy,         w11 = fx * fy;
    const bool vx0 = (x0 >= 0) && (x0 <= 127), vx1 = (x0 + 1 >= 0) && (x0 + 1 <= 127);
    const bool vy0 = (y0 >= 0) && (y0 <= 127), vy1 = (y0 + 1 >= 0) && (y0 + 1 <= 127);
    float wsum = 0.f;
    if (vx0 && vy0) { acc += ptw * w00 * fb[y0 * 128 + x0];           wsum += w00; }
    if (vx1 && vy0) { acc += ptw * w10 * fb[y0 * 128 + x0 + 1];       wsum += w10; }
    if (vx0 && vy1) { acc += ptw * w01 * fb[(y0 + 1) * 128 + x0];     wsum += w01; }
    if (vx1 && vy1) { acc += ptw * w11 * fb[(y0 + 1) * 128 + x0 + 1]; wsum += w11; }
    stot += ptw * wsum;
  }
  h[(size_t)row * 256 + tid] = acc;
  if (tid == 0) s[row] = stot;
}

// out[n] = v[K] . W[K,N] column n  (for bev_value_b @ bev_out_w)
__global__ void vecmat_kernel(const float* __restrict__ v, const float* __restrict__ W,
                              int K, int N, float* __restrict__ out) {
  const int n = blockIdx.x * blockDim.x + threadIdx.x;
  if (n >= N) return;
  float a = 0.f;
  for (int k = 0; k < K; ++k) a += v[k] * W[(size_t)k * N + n];
  out[n] = a;
}

// T = traj + proj + s[row]*bvbp[c]
__global__ void add_bev_kernel(const float* __restrict__ traj, const float* __restrict__ proj,
                               const float* __restrict__ s, const float* __restrict__ bvbp,
                               float* __restrict__ T) {
  const int i = blockIdx.x * 256 + threadIdx.x;
  const int row = i >> 8, c = i & 255;
  T[i] = traj[i] + proj[i] + s[row] * bvbp[c];
}

// Agents MHA core: heads=8, dh=32, Lk=32 == wave32. One wave per (b,h,q).
__global__ void attn_kernel(const float* __restrict__ Q, const float* __restrict__ Kb,
                            const float* __restrict__ Vb, float* __restrict__ O) {
  const int lane = threadIdx.x & 31;
  const int wid  = threadIdx.x >> 5;
  const int g = blockIdx.x * 8 + wid;          // 32*8*20 = 5120 waves
  if (g >= 5120) return;
  const int b = g / 160, rem = g % 160, hh = rem / 20, q = rem % 20;
  const float* qp = Q + ((size_t)(b * 20 + q)) * 256 + hh * 32;
  const float* kp = Kb + ((size_t)(b * 32 + lane)) * 256 + hh * 32;
  float dot = 0.f;
  #pragma unroll
  for (int d = 0; d < 32; ++d) dot += qp[d] * kp[d];
  dot *= 0.1767766952966369f;                  // 1/sqrt(32)
  float m = dot;
  for (int off = 16; off; off >>= 1) m = fmaxf(m, __shfl_xor(m, off, 32));
  const float e = __expf(dot - m);
  float se = e;
  for (int off = 16; off; off >>= 1) se += __shfl_xor(se, off, 32);
  const float a = e / se;
  const float* vp = Vb + ((size_t)(b * 32)) * 256 + hh * 32 + lane;
  float out = 0.f;
  #pragma unroll 8
  for (int k = 0; k < 32; ++k) out += __shfl(a, k, 32) * vp[(size_t)k * 256];
  O[((size_t)(b * 20 + q)) * 256 + hh * 32 + lane] = out;
}

// LayerNorm over last dim 256. Optional additive input with row mapping row/addDiv.
__global__ void ln_kernel(const float* __restrict__ X, const float* __restrict__ Add,
                          int addDiv, const float* __restrict__ g,
                          const float* __restrict__ bta, float* __restrict__ Y) {
  const int row = blockIdx.x, tid = threadIdx.x;
  __shared__ float red[256];
  float x = X[(size_t)row * 256 + tid];
  if (Add) x += Add[(size_t)(row / addDiv) * 256 + tid];
  red[tid] = x; __syncthreads();
  for (int s = 128; s > 0; s >>= 1) { if (tid < s) red[tid] += red[tid + s]; __syncthreads(); }
  const float m = red[0] * (1.f / 256.f);
  __syncthreads();
  const float d = x - m;
  red[tid] = d * d; __syncthreads();
  for (int s = 128; s > 0; s >>= 1) { if (tid < s) red[tid] += red[tid + s]; __syncthreads(); }
  const float v = red[0] * (1.f / 256.f);
  Y[(size_t)row * 256 + tid] = d * rsqrtf(v + 1e-5f) * g[tid] + bta[tid];
}

__global__ void mish_kernel(const float* __restrict__ x, float* __restrict__ y, int n) {
  const int i = blockIdx.x * 256 + threadIdx.x;
  if (i >= n) return;
  const float v = x[i];
  const float sp = (v > 15.f) ? v : log1pf(__expf(v));
  y[i] = v * tanhf(sp);
}

// T = T*(1+scale) + shift, scale/shift = ss[b, 0:256] / ss[b, 256:512], b = row/20
__global__ void modulate_kernel(const float* __restrict__ ss, float* __restrict__ T) {
  const int i = blockIdx.x * 256 + threadIdx.x;
  const int row = i >> 8, c = i & 255;
  const int b = row / 20;
  T[i] = T[i] * (1.f + ss[(size_t)b * 512 + c]) + ss[(size_t)b * 512 + 256 + c];
}

// plan_cls[row] = X[row,:] . w + b0   (one wave per row)
__global__ void clsdot_kernel(const float* __restrict__ X, const float* __restrict__ w,
                              const float* __restrict__ b, float* __restrict__ out) {
  const int lane = threadIdx.x & 31, wid = threadIdx.x >> 5;
  const int row = blockIdx.x * 8 + wid;
  if (row >= 640) return;
  float a = 0.f;
  for (int c = lane; c < 256; c += 32) a += X[(size_t)row * 256 + c] * w[c];
  for (int off = 16; off; off >>= 1) a += __shfl_xor(a, off, 32);
  if (lane == 0) out[row] = a + b[0];
}

// poses_reg: xy = r3[:2] + noisy, heading = tanh(r3[2]) * pi
__global__ void poses_kernel(const float* __restrict__ r3, const float* __restrict__ noisy,
                             float* __restrict__ out) {
  const int i = blockIdx.x * 256 + threadIdx.x;     // (row, t): 640*8
  if (i >= 5120) return;
  const int row = i >> 3, t = i & 7;
  const float rx = r3[(size_t)row * 24 + t * 3 + 0];
  const float ry = r3[(size_t)row * 24 + t * 3 + 1];
  const float rh = r3[(size_t)row * 24 + t * 3 + 2];
  out[(size_t)i * 3 + 0] = rx + noisy[(size_t)i * 2 + 0];
  out[(size_t)i * 3 + 1] = ry + noisy[(size_t)i * 2 + 1];
  out[(size_t)i * 3 + 2] = tanhf(rh) * 3.14159265358979323846f;
}

// ---------------------------------------------------------------------------
extern "C" void kernel_launch(void* const* d_in, const int* in_sizes, int n_in,
                              void* d_out, int out_size, void* d_ws, size_t ws_size,
                              hipStream_t stream) {
  const float* traj_feature = (const float*)d_in[0];
  const float* noisy        = (const float*)d_in[1];
  const float* bev_feature  = (const float*)d_in[2];
  const float* agents_query = (const float*)d_in[3];
  const float* ego_query    = (const float*)d_in[4];
  const float* time_embed   = (const float*)d_in[5];
  // d_in[6] status_encoding: unused by the reference forward
  const float* bev_value_w  = (const float*)d_in[7];
  const float* bev_value_b  = (const float*)d_in[8];
  const float* bev_attn_w   = (const float*)d_in[9];
  const float* bev_attn_b   = (const float*)d_in[10];
  const float* bev_out_w    = (const float*)d_in[11];
  const float* bev_out_b    = (const float*)d_in[12];
  const float* aq_wq = (const float*)d_in[13]; const float* aq_bq = (const float*)d_in[14];
  const float* aq_wk = (const float*)d_in[15]; const float* aq_bk = (const float*)d_in[16];
  const float* aq_wv = (const float*)d_in[17]; const float* aq_bv = (const float*)d_in[18];
  const float* aq_wo = (const float*)d_in[19]; const float* aq_bo = (const float*)d_in[20];
  // eq_wq(21)/eq_bq(22)/eq_wk(23)/eq_bk(24) unused: softmax over single key == 1
  const float* eq_wv = (const float*)d_in[25]; const float* eq_bv = (const float*)d_in[26];
  const float* eq_wo = (const float*)d_in[27]; const float* eq_bo = (const float*)d_in[28];
  const float* ffn_w1 = (const float*)d_in[29]; const float* ffn_b1 = (const float*)d_in[30];
  const float* ffn_w2 = (const float*)d_in[31]; const float* ffn_b2 = (const float*)d_in[32];
  const float* n1_g = (const float*)d_in[33]; const float* n1_b = (const float*)d_in[34];
  const float* n2_g = (const float*)d_in[35]; const float* n2_b = (const float*)d_in[36];
  const float* n3_g = (const float*)d_in[37]; const float* n3_b = (const float*)d_in[38];
  const float* mod_w = (const float*)d_in[39]; const float* mod_b = (const float*)d_in[40];
  const float* cls_w1 = (const float*)d_in[41]; const float* cls_b1 = (const float*)d_in[42];
  const float* cls_ln1_g = (const float*)d_in[43]; const float* cls_ln1_b = (const float*)d_in[44];
  const float* cls_w2 = (const float*)d_in[45]; const float* cls_b2 = (const float*)d_in[46];
  const float* cls_ln2_g = (const float*)d_in[47]; const float* cls_ln2_b = (const float*)d_in[48];
  const float* cls_w3 = (const float*)d_in[49]; const float* cls_b3 = (const float*)d_in[50];
  const float* reg_w1 = (const float*)d_in[51]; const float* reg_b1 = (const float*)d_in[52];
  const float* reg_w2 = (const float*)d_in[53]; const float* reg_b2 = (const float*)d_in[54];
  const float* reg_w3 = (const float*)d_in[55]; const float* reg_b3 = (const float*)d_in[56];

  float* out = (float*)d_out;      // [0,15360): poses_reg ; [15360,16000): plan_cls

  // workspace carve-up (floats)
  float* ws = (float*)d_ws;
  size_t off = 0;
  auto alloc = [&](size_t n) { float* p = ws + off; off += n; return p; };
  float* hbuf = alloc(640 * 256);
  float* sbuf = alloc(640);
  float* bvbp = alloc(256);
  float* tmpA = alloc(640 * 256);
  float* tmpB = alloc(640 * 256);
  float* T    = alloc(640 * 256);
  float* qb   = alloc(640 * 256);
  float* kb   = alloc(1024 * 256);
  float* vb   = alloc(1024 * 256);
  float* ob   = alloc(640 * 256);
  float* f1   = alloc(640 * 1024);
  float* msh  = alloc(32 * 256);
  float* ssb  = alloc(32 * 512);
  float* r3   = alloc(640 * 24);

  auto gemm = [&](const float* A, const float* W, const float* bias, float* C,
                  int M, int N, int K, int relu) {
    const int ntiles = (M / 16) * ((N + 15) / 16);
    const int blocks = (ntiles + 7) / 8;
    gemm_bf16_wmma<<<blocks, 256, 0, stream>>>(A, W, bias, C, M, N, K, relu);
  };

  // --- BEV cross-attention (algebraically fused: gather first, project 640 rows only)
  bev_gather_kernel<<<640, 256, 0, stream>>>(traj_feature, noisy, bev_feature,
                                             bev_attn_w, bev_attn_b, hbuf, sbuf);
  vecmat_kernel<<<1, 256, 0, stream>>>(bev_value_b, bev_out_w, 256, 256, bvbp);
  gemm(hbuf, bev_value_w, nullptr, tmpA, 640, 256, 256, 0);
  gemm(tmpA, bev_out_w, bev_out_b, tmpB, 640, 256, 256, 0);
  add_bev_kernel<<<640, 256, 0, stream>>>(traj_feature, tmpB, sbuf, bvbp, T);

  // --- agents MHA + LN1
  gemm(T, aq_wq, aq_bq, qb, 640, 256, 256, 0);
  gemm(agents_query, aq_wk, aq_bk, kb, 1024, 256, 256, 0);
  gemm(agents_query, aq_wv, aq_bv, vb, 1024, 256, 256, 0);
  attn_kernel<<<640, 256, 0, stream>>>(qb, kb, vb, ob);
  gemm(ob, aq_wo, aq_bo, tmpA, 640, 256, 256, 0);
  ln_kernel<<<640, 256, 0, stream>>>(T, tmpA, 1, n1_g, n1_b, T);

  // --- ego MHA (Lk==1 => softmax==1 => output = (ego@Wv+bv)@Wo+bo broadcast) + LN2
  gemm(ego_query, eq_wv, eq_bv, tmpA, 32, 256, 256, 0);
  gemm(tmpA, eq_wo, eq_bo, tmpB, 32, 256, 256, 0);
  ln_kernel<<<640, 256, 0, stream>>>(T, tmpB, 20, n2_g, n2_b, T);

  // --- FFN + LN3 (no residual per reference)
  gemm(T, ffn_w1, ffn_b1, f1, 640, 1024, 256, 1);
  gemm(f1, ffn_w2, ffn_b2, tmpA, 640, 256, 1024, 0);
  ln_kernel<<<640, 256, 0, stream>>>(tmpA, nullptr, 1, n3_g, n3_b, T);

  // --- time modulation
  mish_kernel<<<32, 256, 0, stream>>>(time_embed, msh, 32 * 256);
  gemm(msh, mod_w, mod_b, ssb, 32, 512, 256, 0);
  modulate_kernel<<<640, 256, 0, stream>>>(ssb, T);

  // --- cls head
  gemm(T, cls_w1, cls_b1, tmpA, 640, 256, 256, 1);
  ln_kernel<<<640, 256, 0, stream>>>(tmpA, nullptr, 1, cls_ln1_g, cls_ln1_b, tmpB);
  gemm(tmpB, cls_w2, cls_b2, tmpA, 640, 256, 256, 1);
  ln_kernel<<<640, 256, 0, stream>>>(tmpA, nullptr, 1, cls_ln2_g, cls_ln2_b, tmpB);
  clsdot_kernel<<<80, 256, 0, stream>>>(tmpB, cls_w3, cls_b3, out + 15360);

  // --- reg head
  gemm(T, reg_w1, reg_b1, tmpA, 640, 256, 256, 1);
  gemm(tmpA, reg_w2, reg_b2, qb, 640, 256, 256, 1);
  gemm(qb, reg_w3, reg_b3, r3, 640, 24, 256, 0);
  poses_kernel<<<20, 256, 0, stream>>>(r3, noisy, out);
}
